// DilatedTransformerDecoderLayer_49916109914350
// MI455X (gfx1250) — matrix operand order repaired and checked by reading, not compiled, not run on hardware
//
#include <hip/hip_runtime.h>

// ---------------------------------------------------------------------------
// DilatedTransformerDecoderLayer for MI455X (gfx1250, wave32, WMMA).
// All GEMM-shaped math runs through v_wmma_f32_16x16x32_bf16 (K=32 per issue,
// fp32 accumulate). LayerNorm/softmax stay fp32. Attention uses a flash-style
// streaming softmax; each dilation group is a 2048x2048x64 block problem with
// memory stride rate*768.
// GEMM wave tile: 32(M) x 64(N) -> 8 WMMA per 32-K step, B fragments reused
// across two A fragments (arithmetic intensity ~21 FLOP/B from cache).
// ---------------------------------------------------------------------------

#define D_MODEL   768
#define NHEAD     12
#define HEAD_DIM  64
#define DIM_FF    3072
#define BATCH     2
#define SEQ       8192
#define BN        (BATCH * SEQ)      // 16384 tokens
#define SR        2048               // subsampled rows per attention block (all groups)
#define GH        4                  // heads per dilation group
#define ATT_SCALE 0.125f             // 1/sqrt(64)
#define LN_EPS    1e-5f

typedef __attribute__((ext_vector_type(16))) __bf16 bf16x16;
typedef __attribute__((ext_vector_type(8)))  float  floatx8;

union Frag { unsigned u[8]; bf16x16 v; };

__device__ __forceinline__ unsigned short f32_to_bf16(float f) {
  unsigned u = __float_as_uint(f);
  u += 0x7FFFu + ((u >> 16) & 1u);     // round-to-nearest-even
  return (unsigned short)(u >> 16);
}

__device__ __forceinline__ floatx8 zero8() {
  floatx8 z = {0.f, 0.f, 0.f, 0.f, 0.f, 0.f, 0.f, 0.f};
  return z;
}

// ---------------------------------------------------------------------------
// fp32 -> bf16 elementwise convert
// ---------------------------------------------------------------------------
__global__ void cvt_bf16_kernel(const float* __restrict__ in,
                                unsigned short* __restrict__ out, int n) {
  int i = blockIdx.x * 256 + threadIdx.x;
  if (i < n) out[i] = f32_to_bf16(in[i]);
}

// ---------------------------------------------------------------------------
// LayerNorm (fp32 in, bf16 out). One block of 256 threads per row.
// ---------------------------------------------------------------------------
__global__ void ln_kernel(const float* __restrict__ x, const float* __restrict__ g,
                          const float* __restrict__ b, unsigned short* __restrict__ out,
                          int D) {
  const int row = blockIdx.x;
  const float* xr = x + (size_t)row * D;
  __shared__ float red[256];

  float s = 0.f;
  for (int i = threadIdx.x; i < D; i += 256) s += xr[i];
  red[threadIdx.x] = s; __syncthreads();
  for (int st = 128; st > 0; st >>= 1) {
    if (threadIdx.x < st) red[threadIdx.x] += red[threadIdx.x + st];
    __syncthreads();
  }
  const float mean = red[0] / (float)D;
  __syncthreads();

  float s2 = 0.f;
  for (int i = threadIdx.x; i < D; i += 256) { float d = xr[i] - mean; s2 += d * d; }
  red[threadIdx.x] = s2; __syncthreads();
  for (int st = 128; st > 0; st >>= 1) {
    if (threadIdx.x < st) red[threadIdx.x] += red[threadIdx.x + st];
    __syncthreads();
  }
  const float inv = rsqrtf(red[0] / (float)D + LN_EPS);

  unsigned short* orow = out + (size_t)row * D;
  for (int i = threadIdx.x; i < D; i += 256)
    orow[i] = f32_to_bf16((xr[i] - mean) * inv * g[i] + b[i]);
}

// ---------------------------------------------------------------------------
// GEMM: Y[M,N] = X[M,K] (bf16) @ W[N,K]^T (bf16) + bias, optional ReLU.
// One wave -> 32x64 output tile: 2 A fragments x 4 B fragments = 8 WMMA per
// 32-K step; each B fragment is reused by both A fragments.
// blockDim = 128 (4 waves).
// ---------------------------------------------------------------------------
__global__ void __launch_bounds__(128)
gemm_bf16_kernel(const unsigned short* __restrict__ X,
                 const unsigned short* __restrict__ W,
                 const float* __restrict__ bias,
                 float* __restrict__ outF,
                 unsigned short* __restrict__ outB,
                 int M, int N, int K, int relu) {
  const int lane = threadIdx.x & 31;
  const int half = lane >> 4;
  const int lm   = lane & 15;
  const int waveId = blockIdx.x * 4 + (threadIdx.x >> 5);
  const int ntiles64 = N >> 6;
  const int mtile = waveId / ntiles64;
  const int ntile = waveId % ntiles64;
  if (mtile * 32 >= M) return;
  const int m0 = mtile * 32;
  const int n0 = ntile * 64;

  floatx8 acc[2][4];
#pragma unroll
  for (int r = 0; r < 2; ++r)
#pragma unroll
    for (int j = 0; j < 4; ++j) acc[r][j] = zero8();

  const unsigned short* Arow0 = X + (size_t)(m0 + lm) * K;
  const unsigned short* Arow1 = X + (size_t)(m0 + 16 + lm) * K;

  for (int k0 = 0; k0 < K; k0 += 32) {
    Frag af[2];
#pragma unroll
    for (int i = 0; i < 4; ++i) {
      const int ka = k0 + half * 8 + 2 * i;
      af[0].u[i]     = *(const unsigned*)(Arow0 + ka);
      af[0].u[4 + i] = *(const unsigned*)(Arow0 + ka + 16);
      af[1].u[i]     = *(const unsigned*)(Arow1 + ka);
      af[1].u[4 + i] = *(const unsigned*)(Arow1 + ka + 16);
    }
#pragma unroll
    for (int j = 0; j < 4; ++j) {
      const unsigned short* Wrow = W + (size_t)(n0 + j * 16 + lm) * K;
      Frag bfr;
#pragma unroll
      for (int jj = 0; jj < 8; ++jj)
        bfr.u[jj] = *(const unsigned*)(Wrow + k0 + half * 8 + (jj >> 2) * 16 + 2 * (jj & 3));
      acc[0][j] = __builtin_amdgcn_wmma_f32_16x16x32_bf16(
          false, af[0].v, false, bfr.v, (short)0, acc[0][j], false, false);
      acc[1][j] = __builtin_amdgcn_wmma_f32_16x16x32_bf16(
          false, af[1].v, false, bfr.v, (short)0, acc[1][j], false, false);
    }
  }

#pragma unroll
  for (int r = 0; r < 2; ++r) {
#pragma unroll
    for (int j = 0; j < 4; ++j) {
      const int n = n0 + j * 16 + lm;
      const float bv = bias ? bias[n] : 0.f;
#pragma unroll
      for (int p = 0; p < 8; ++p) {
        const int m = m0 + r * 16 + half * 8 + p;
        float y = acc[r][j][p] + bv;
        if (relu) y = fmaxf(y, 0.f);
        const size_t o = (size_t)m * N + n;
        if (outF) outF[o] = y;
        if (outB) outB[o] = f32_to_bf16(y);
      }
    }
  }
}

// ---------------------------------------------------------------------------
// Dilated flash attention. Q/K/V are bf16 [B, SEQ, NHEAD, HEAD_DIM].
// Grid: 56 blocks (group/batch/segment/head) x 32 q-tiles = 1792 workgroups,
// 128 threads (4 waves, 16 q-rows each). Output fp32, pre-zeroed, scaled 1/3.
// ---------------------------------------------------------------------------
__global__ void __launch_bounds__(128)
dilated_attn_kernel(const unsigned short* __restrict__ Q,
                    const unsigned short* __restrict__ K,
                    const unsigned short* __restrict__ V,
                    float* __restrict__ O,
                    const int* __restrict__ causal_flag) {
  const int lane = threadIdx.x & 31;
  const int wave = threadIdx.x >> 5;
  const int half = lane >> 4;
  const int lm   = lane & 15;

  const int qt  = blockIdx.x & 31;
  const int blk = blockIdx.x >> 5;
  int group, bb, seg, hh;
  if (blk < 32)      { group = 0; bb = blk >> 4;            seg = (blk >> 2) & 3; hh = blk & 3; }
  else if (blk < 48) { int t = blk - 32; group = 1; bb = t >> 3; seg = (t >> 2) & 1; hh = t & 3; }
  else               { int t = blk - 48; group = 2; bb = t >> 2; seg = 0;            hh = t & 3; }
  const int segl = (group == 0) ? 2048 : (group == 1) ? 4096 : 8192;
  const int rate = 1 << group;                                  // 1,2,4
  const int offs = (group == 0) ? 0 : (group == 1) ? 1 : 2;     // i % r
  const int head = group * GH + hh;
  const int causal = causal_flag[0];

  const size_t rs   = (size_t)D_MODEL;
  const size_t tok0 = (size_t)bb * SEQ + (size_t)seg * segl + offs;
  const size_t base = tok0 * rs + (size_t)head * HEAD_DIM;
  const size_t rstr = (size_t)rate * rs;      // element stride between sub-rows

  const int qbase = qt * 64 + wave * 16;

  // Q A-fragments (hd 0..31 and 32..63), kept in registers all kernel
  const unsigned short* qrow = Q + base + (size_t)(qbase + lm) * rstr;
  Frag qf[2];
#pragma unroll
  for (int c = 0; c < 2; ++c)
#pragma unroll
    for (int i = 0; i < 4; ++i) {
      const int ka = c * 32 + half * 8 + 2 * i;
      qf[c].u[i]     = *(const unsigned*)(qrow + ka);
      qf[c].u[4 + i] = *(const unsigned*)(qrow + ka + 16);
    }

  floatx8 oacc[4];
#pragma unroll
  for (int j = 0; j < 4; ++j) oacc[j] = zero8();
  float rmax[8], rsum[8];
#pragma unroll
  for (int p = 0; p < 8; ++p) { rmax[p] = -1e30f; rsum[p] = 0.f; }

  __shared__ __align__(16) unsigned short plds[4][16][32];
  const int qmaxrow = qbase + 15;

  for (int kc = 0; kc < SR; kc += 32) {
    if (causal && kc > qmaxrow) break;

    // ---- S = Q K^T (two 16x16 column tiles) ----
    floatx8 sf[2];
#pragma unroll
    for (int f = 0; f < 2; ++f) {
      const unsigned short* krow = K + base + (size_t)(kc + f * 16 + lm) * rstr;
      floatx8 cacc = zero8();
#pragma unroll
      for (int c = 0; c < 2; ++c) {
        Frag bfr;
#pragma unroll
        for (int jj = 0; jj < 8; ++jj)
          bfr.u[jj] = *(const unsigned*)(krow + c * 32 + half * 8 + (jj >> 2) * 16 + 2 * (jj & 3));
        cacc = __builtin_amdgcn_wmma_f32_16x16x32_bf16(
            false, qf[c].v, false, bfr.v, (short)0, cacc, false, false);
      }
      sf[f] = cacc;
    }

    // ---- online softmax (row stats replicated across the 16-lane half-group) ----
#pragma unroll
    for (int p = 0; p < 8; ++p) {
      const int qidx = qbase + half * 8 + p;
      float v0 = sf[0][p] * ATT_SCALE;
      float v1 = sf[1][p] * ATT_SCALE;
      if (causal) {
        if (kc + lm > qidx)      v0 = -1e9f;
        if (kc + 16 + lm > qidx) v1 = -1e9f;
      }
      float mx = fmaxf(v0, v1);
#pragma unroll
      for (int sh = 1; sh < 16; sh <<= 1) mx = fmaxf(mx, __shfl_xor(mx, sh, 32));
      const float mnew = fmaxf(rmax[p], mx);
      const float corr = __expf(rmax[p] - mnew);
      const float e0 = __expf(v0 - mnew);
      const float e1 = __expf(v1 - mnew);
      float ls = e0 + e1;
#pragma unroll
      for (int sh = 1; sh < 16; sh <<= 1) ls += __shfl_xor(ls, sh, 32);
      rsum[p] = rsum[p] * corr + ls;
      rmax[p] = mnew;
#pragma unroll
      for (int j = 0; j < 4; ++j) oacc[j][p] *= corr;
      // stash probabilities (C layout) into the wave's LDS tile as bf16
      const int row = half * 8 + p;
      plds[wave][row][lm]      = f32_to_bf16(e0);
      plds[wave][row][16 + lm] = f32_to_bf16(e1);
    }
    asm volatile("s_wait_dscnt 0x0" ::: "memory");

    // reload P in A-fragment layout
    Frag pf;
#pragma unroll
    for (int i = 0; i < 4; ++i) {
      const int ka = half * 8 + 2 * i;
      pf.u[i]     = *(const unsigned*)&plds[wave][lm][ka];
      pf.u[4 + i] = *(const unsigned*)&plds[wave][lm][ka + 16];
    }

    // ---- O += P V ----
#pragma unroll
    for (int j = 0; j < 4; ++j) {
      Frag vfr;
#pragma unroll
      for (int jj = 0; jj < 8; ++jj) {
        const int k0 = kc + half * 8 + (jj >> 2) * 16 + 2 * (jj & 3);
        const unsigned short* vp = V + base + (size_t)k0 * rstr + j * 16 + lm;
        const unsigned lo = *vp;
        const unsigned hi = *(vp + rstr);
        vfr.u[jj] = lo | (hi << 16);
      }
      oacc[j] = __builtin_amdgcn_wmma_f32_16x16x32_bf16(
          false, pf.v, false, vfr.v, (short)0, oacc[j], false, false);
    }
  }

  // ---- epilogue: normalize, /num_groups, scatter ----
#pragma unroll
  for (int p = 0; p < 8; ++p) {
    const int qidx = qbase + half * 8 + p;
    const float inv = (1.0f / 3.0f) / rsum[p];
    float* orow = O + (tok0 + (size_t)qidx * rate) * rs + (size_t)head * HEAD_DIM;
#pragma unroll
    for (int j = 0; j < 4; ++j)
      orow[j * 16 + lm] = oacc[j][p] * inv;
  }
}

// ---------------------------------------------------------------------------
// Host-side orchestration
// ---------------------------------------------------------------------------
static inline void launch_gemm(const unsigned short* X, const unsigned short* W,
                               const float* bias, float* outF, unsigned short* outB,
                               int M, int N, int K, int relu, hipStream_t s) {
  const int waves = (M / 32) * (N / 64);
  gemm_bf16_kernel<<<waves / 4, 128, 0, s>>>(X, W, bias, outF, outB, M, N, K, relu);
}
static inline void launch_cvt(const float* src, unsigned short* dst, int n, hipStream_t s) {
  cvt_bf16_kernel<<<(n + 255) / 256, 256, 0, s>>>(src, dst, n);
}

extern "C" void kernel_launch(void* const* d_in, const int* in_sizes, int n_in,
                              void* d_out, int out_size, void* d_ws, size_t ws_size,
                              hipStream_t stream) {
  (void)in_sizes; (void)n_in; (void)out_size; (void)ws_size;

  const float* tgt    = (const float*)d_in[0];
  const float* memory = (const float*)d_in[1];
  const float* sa_wq = (const float*)d_in[2];  const float* sa_bq = (const float*)d_in[3];
  const float* sa_wk = (const float*)d_in[4];  const float* sa_bk = (const float*)d_in[5];
  const float* sa_wv = (const float*)d_in[6];  const float* sa_bv = (const float*)d_in[7];
  const float* sa_wo = (const float*)d_in[8];  const float* sa_bo = (const float*)d_in[9];
  const float* sa_lg = (const float*)d_in[10]; const float* sa_lb = (const float*)d_in[11];
  const float* ca_wq = (const float*)d_in[12]; const float* ca_bq = (const float*)d_in[13];
  const float* ca_wk = (const float*)d_in[14]; const float* ca_bk = (const float*)d_in[15];
  const float* ca_wv = (const float*)d_in[16]; const float* ca_bv = (const float*)d_in[17];
  const float* ca_wo = (const float*)d_in[18]; const float* ca_bo = (const float*)d_in[19];
  const float* ca_lg = (const float*)d_in[20]; const float* ca_lb = (const float*)d_in[21];
  const float* n1_g  = (const float*)d_in[22]; const float* n1_b  = (const float*)d_in[23];
  const float* n2_g  = (const float*)d_in[24]; const float* n2_b  = (const float*)d_in[25];
  const float* n3_g  = (const float*)d_in[26]; const float* n3_b  = (const float*)d_in[27];
  const float* n4_g  = (const float*)d_in[28]; const float* n4_b  = (const float*)d_in[29];
  const float* l1_w  = (const float*)d_in[30]; const float* l1_b  = (const float*)d_in[31];
  const float* l2_w  = (const float*)d_in[32]; const float* l2_b  = (const float*)d_in[33];
  const int* tgt_causal = (const int*)d_in[34];
  const int* mem_causal = (const int*)d_in[35];
  float* out = (float*)d_out;

  // Workspace carve (~550 MB total)
  char* base = (char*)d_ws; size_t off = 0;
#define TAKE(var, type, count)                                            \
  type* var = (type*)(base + off);                                        \
  off += (((size_t)(count) * sizeof(type)) + 255) & ~(size_t)255;
  TAKE(wq1, unsigned short, 768 * 768)  TAKE(wk1, unsigned short, 768 * 768)
  TAKE(wv1, unsigned short, 768 * 768)  TAKE(wo1, unsigned short, 768 * 768)
  TAKE(wq2, unsigned short, 768 * 768)  TAKE(wk2, unsigned short, 768 * 768)
  TAKE(wv2, unsigned short, 768 * 768)  TAKE(wo2, unsigned short, 768 * 768)
  TAKE(w1b, unsigned short, (size_t)DIM_FF * 768)
  TAKE(w2b, unsigned short, (size_t)768 * DIM_FF)
  TAKE(bfA,  unsigned short, (size_t)BN * D_MODEL)   // LN outputs (bf16)
  TAKE(bfM,  unsigned short, (size_t)BN * D_MODEL)   // memory in bf16
  TAKE(qb,   unsigned short, (size_t)BN * D_MODEL)
  TAKE(kb,   unsigned short, (size_t)BN * D_MODEL)
  TAKE(vb,   unsigned short, (size_t)BN * D_MODEL)
  TAKE(attnF, float,         (size_t)BN * D_MODEL)   // attention scatter buffer
  TAKE(bufX,  float,         (size_t)BN * D_MODEL)   // mhda outputs
  TAKE(ffnF,  float,         (size_t)BN * DIM_FF)    // relu(l1) fp32
  TAKE(ffnB,  unsigned short,(size_t)BN * DIM_FF)    // LN4 output bf16
#undef TAKE

  const int WSZ = 768 * 768;
  launch_cvt(sa_wq, wq1, WSZ, stream); launch_cvt(sa_wk, wk1, WSZ, stream);
  launch_cvt(sa_wv, wv1, WSZ, stream); launch_cvt(sa_wo, wo1, WSZ, stream);
  launch_cvt(ca_wq, wq2, WSZ, stream); launch_cvt(ca_wk, wk2, WSZ, stream);
  launch_cvt(ca_wv, wv2, WSZ, stream); launch_cvt(ca_wo, wo2, WSZ, stream);
  launch_cvt(l1_w, w1b, DIM_FF * 768, stream);
  launch_cvt(l2_w, w2b, 768 * DIM_FF, stream);
  launch_cvt(memory, bfM, BN * D_MODEL, stream);

  // ---- self attention branch ----
  ln_kernel<<<BN, 256, 0, stream>>>(tgt, n1_g, n1_b, bfA, D_MODEL);
  launch_gemm(bfA, wq1, sa_bq, nullptr, qb, BN, D_MODEL, D_MODEL, 0, stream);
  launch_gemm(bfA, wk1, sa_bk, nullptr, kb, BN, D_MODEL, D_MODEL, 0, stream);
  launch_gemm(bfA, wv1, sa_bv, nullptr, vb, BN, D_MODEL, D_MODEL, 0, stream);
  hipMemsetAsync(attnF, 0, (size_t)BN * D_MODEL * sizeof(float), stream);
  dilated_attn_kernel<<<1792, 128, 0, stream>>>(qb, kb, vb, attnF, tgt_causal);
  ln_kernel<<<BN, 256, 0, stream>>>(attnF, sa_lg, sa_lb, bfA, D_MODEL);
  launch_gemm(bfA, wo1, sa_bo, bufX, nullptr, BN, D_MODEL, D_MODEL, 0, stream);

  // ---- cross attention branch ----
  ln_kernel<<<BN, 256, 0, stream>>>(bufX, n2_g, n2_b, bfA, D_MODEL);
  launch_gemm(bfA, wq2, ca_bq, nullptr, qb, BN, D_MODEL, D_MODEL, 0, stream);
  launch_gemm(bfM, wk2, ca_bk, nullptr, kb, BN, D_MODEL, D_MODEL, 0, stream);
  launch_gemm(bfM, wv2, ca_bv, nullptr, vb, BN, D_MODEL, D_MODEL, 0, stream);
  hipMemsetAsync(attnF, 0, (size_t)BN * D_MODEL * sizeof(float), stream);
  dilated_attn_kernel<<<1792, 128, 0, stream>>>(qb, kb, vb, attnF, mem_causal);
  ln_kernel<<<BN, 256, 0, stream>>>(attnF, ca_lg, ca_lb, bfA, D_MODEL);
  launch_gemm(bfA, wo2, ca_bo, bufX, nullptr, BN, D_MODEL, D_MODEL, 0, stream);

  // ---- FFN ----
  ln_kernel<<<BN, 256, 0, stream>>>(bufX, n3_g, n3_b, bfA, D_MODEL);
  launch_gemm(bfA, w1b, l1_b, ffnF, nullptr, BN, DIM_FF, D_MODEL, 1, stream);
  ln_kernel<<<BN, 256, 0, stream>>>(ffnF, n4_g, n4_b, ffnB, DIM_FF);
  launch_gemm(ffnB, w2b, l2_b, out, nullptr, BN, D_MODEL, DIM_FF, 0, stream);
}
